// ARSSMHyperbolicFusion_74053826118137
// MI455X (gfx1250) — matrix operand order repaired and checked by reading, not compiled, not run on hardware
//
#include <hip/hip_runtime.h>
#include <hip/hip_bf16.h>
#include <math.h>

// Problem constants (match reference)
#define DM   1024      // d_model
#define DS   64        // d_state
#define RB   4         // batch
#define SS   2048      // seq len
#define TOK  (RB*SS)   // 8192 tokens
#define TCH  16        // scan chunk length
#define NCH  (SS/TCH)  // 128 chunks per batch
#define EPSV 1e-6f

typedef __attribute__((ext_vector_type(16))) _Float16 v16h;
typedef __attribute__((ext_vector_type(8)))  float    v8f;
typedef __attribute__((ext_vector_type(4)))  unsigned int v4u;
typedef __attribute__((ext_vector_type(8)))  int      v8i;
typedef __attribute__((ext_vector_type(4)))  int      v4i;

__device__ __forceinline__ float sigmoidf_(float v) { return 1.0f / (1.0f + __expf(-v)); }

// ---------------------------------------------------------------------------
// Tensor Data Mover: 1-D contiguous f32 copy global -> LDS (ISA ch.7/8 D#).
// group0: count=1 | lds_addr | global_addr(57b) | type=2("image")
// group1: data_size=2(4B), tensor_dim0=tile_dim0=nelem, stride=nelem, 1-D tile.
// Issue from one wave per block; completion via s_wait_tensorcnt + barrier.
__device__ __forceinline__ void tdm_load_1d(const void* gaddr, unsigned lds_byte_off,
                                            unsigned nelem_f32) {
    unsigned long long ga = (unsigned long long)(uintptr_t)gaddr;
    v4u g0;
    g0[0] = 1u;                                            // count=1, user desc
    g0[1] = lds_byte_off;                                  // lds_addr (bytes)
    g0[2] = (unsigned)(ga & 0xFFFFFFFFu);                  // global_addr[31:0]
    g0[3] = (unsigned)((ga >> 32) & 0x1FFFFFFu)            // global_addr[56:32]
          | (2u << 30);                                    // type=2
    v8i g1;
    g1[0] = (int)(2u << 16);                               // data_size=4B, no mcast
    g1[1] = (int)((nelem_f32 & 0xFFFFu) << 16);            // tensor_dim0[15:0]
    g1[2] = (int)((nelem_f32 >> 16) & 0xFFFFu)             // tensor_dim0[31:16]
          | (1 << 16);                                     // tensor_dim1 = 1
    g1[3] = (int)((nelem_f32 & 0xFFFFu) << 16);            // tile_dim0
    g1[4] = 0;                                             // tile_dim1/2 unused (1-D)
    g1[5] = (int)nelem_f32;                                // tensor_dim0_stride[31:0]
    g1[6] = 0;
    g1[7] = 0;
    v4i z4 = {0, 0, 0, 0};
#if defined(__clang_major__) && (__clang_major__ >= 23)
    v8i z8 = {0, 0, 0, 0, 0, 0, 0, 0};
    __builtin_amdgcn_tensor_load_to_lds(g0, g1, z4, z4, z8, 0);
#else
    __builtin_amdgcn_tensor_load_to_lds(g0, g1, z4, z4, 0);
#endif
}

// The TDM builtin's LDS destination is only encoded inside the descriptor, so
// clang's alias analysis cannot see the __shared__ array being written (reads
// would fold to undef). Publish the LDS pointer through an asm with a memory
// clobber: compiler must now assume this point (the TDM transfer) wrote it.
__device__ __forceinline__ void lds_acquire(const void* p) {
    asm volatile("" : : "v"(p) : "memory");
}

// ---------------------------------------------------------------------------
// WMMA fragment loaders (f32 -> f16 fragments), ISA 7.12.2 layouts.
// A-matrix 16x32 f16: lanes 0-15 row M=lane, K={0..7,16..23}; lanes 16-31 same
// rows, K={8..15,24..31}.
__device__ __forceinline__ v16h load_a_frag(const float* __restrict__ base, int ld, int K0) {
    const int lane = threadIdx.x & 31;
    const int r = lane & 15, h = lane >> 4;
    const float* p = base + (size_t)r * ld + K0 + 8 * h;
    float4 q0 = *(const float4*)(p);
    float4 q1 = *(const float4*)(p + 4);
    float4 q2 = *(const float4*)(p + 16);
    float4 q3 = *(const float4*)(p + 20);
    v16h a;
    a[0]=(_Float16)q0.x;  a[1]=(_Float16)q0.y;  a[2]=(_Float16)q0.z;  a[3]=(_Float16)q0.w;
    a[4]=(_Float16)q1.x;  a[5]=(_Float16)q1.y;  a[6]=(_Float16)q1.z;  a[7]=(_Float16)q1.w;
    a[8]=(_Float16)q2.x;  a[9]=(_Float16)q2.y;  a[10]=(_Float16)q2.z; a[11]=(_Float16)q2.w;
    a[12]=(_Float16)q3.x; a[13]=(_Float16)q3.y; a[14]=(_Float16)q3.z; a[15]=(_Float16)q3.w;
    return a;
}

// Same layout, sourced from LDS (compiles to ds_load_b128 + v_cvt_pk_f16_f32).
__device__ __forceinline__ v16h load_a_frag_lds(const float* base, int ld, int K0) {
    const int lane = threadIdx.x & 31;
    const int r = lane & 15, h = lane >> 4;
    const float* p = base + r * ld + K0 + 8 * h;
    float4 q0 = *(const float4*)(p);
    float4 q1 = *(const float4*)(p + 4);
    float4 q2 = *(const float4*)(p + 16);
    float4 q3 = *(const float4*)(p + 20);
    v16h a;
    a[0]=(_Float16)q0.x;  a[1]=(_Float16)q0.y;  a[2]=(_Float16)q0.z;  a[3]=(_Float16)q0.w;
    a[4]=(_Float16)q1.x;  a[5]=(_Float16)q1.y;  a[6]=(_Float16)q1.z;  a[7]=(_Float16)q1.w;
    a[8]=(_Float16)q2.x;  a[9]=(_Float16)q2.y;  a[10]=(_Float16)q2.z; a[11]=(_Float16)q2.w;
    a[12]=(_Float16)q3.x; a[13]=(_Float16)q3.y; a[14]=(_Float16)q3.z; a[15]=(_Float16)q3.w;
    return a;
}

// B-matrix 32x16 f16: lanes 0-15 hold column n=lane, K={0..7,16..23}; lanes 16-31 hold
// column n=lane-16, K={8..15,24..31}. Source W row-major with W[n][k] = B[k][n].
__device__ __forceinline__ v16h load_b_frag(const float* __restrict__ base, int ld, int K0) {
    const int lane = threadIdx.x & 31;
    const int r = lane & 15, h = lane >> 4;
    const float* p = base + (size_t)r * ld + K0 + 8 * h;
    float4 q0 = *(const float4*)(p);
    float4 q1 = *(const float4*)(p + 4);
    float4 q2 = *(const float4*)(p + 16);
    float4 q3 = *(const float4*)(p + 20);
    v16h b;
    b[0]=(_Float16)q0.x;  b[1]=(_Float16)q0.y;  b[2]=(_Float16)q0.z;  b[3]=(_Float16)q0.w;
    b[4]=(_Float16)q1.x;  b[5]=(_Float16)q1.y;  b[6]=(_Float16)q1.z;  b[7]=(_Float16)q1.w;
    b[8]=(_Float16)q2.x;  b[9]=(_Float16)q2.y;  b[10]=(_Float16)q2.z; b[11]=(_Float16)q2.w;
    b[12]=(_Float16)q3.x; b[13]=(_Float16)q3.y; b[14]=(_Float16)q3.z; b[15]=(_Float16)q3.w;
    return b;
}

__device__ __forceinline__ v16h load_b_frag_lds(const float* base, int ld, int K0) {
    const int lane = threadIdx.x & 31;
    const int r = lane & 15, h = lane >> 4;
    const float* p = base + r * ld + K0 + 8 * h;
    float4 q0 = *(const float4*)(p);
    float4 q1 = *(const float4*)(p + 4);
    float4 q2 = *(const float4*)(p + 16);
    float4 q3 = *(const float4*)(p + 20);
    v16h b;
    b[0]=(_Float16)q0.x;  b[1]=(_Float16)q0.y;  b[2]=(_Float16)q0.z;  b[3]=(_Float16)q0.w;
    b[4]=(_Float16)q1.x;  b[5]=(_Float16)q1.y;  b[6]=(_Float16)q1.z;  b[7]=(_Float16)q1.w;
    b[8]=(_Float16)q2.x;  b[9]=(_Float16)q2.y;  b[10]=(_Float16)q2.z; b[11]=(_Float16)q2.w;
    b[12]=(_Float16)q3.x; b[13]=(_Float16)q3.y; b[14]=(_Float16)q3.z; b[15]=(_Float16)q3.w;
    return b;
}

// ---------------------------------------------------------------------------
// K1: per-token rank weight: ||x|| -> atanh distance -> MLP(1,32,1) -> sigmoid,
// times physics gate. One wave32 per token, 8 waves per block.
__global__ void k_rankw(const float* __restrict__ x, const float* __restrict__ gr,
                        const float* __restrict__ gi, const float* __restrict__ w1,
                        const float* __restrict__ b1, const float* __restrict__ w2,
                        const float* __restrict__ b2, const float* __restrict__ pw,
                        const float* __restrict__ pb, float* __restrict__ rw) {
    const int lane = threadIdx.x & 31;
    const int tok  = blockIdx.x * (blockDim.x >> 5) + (threadIdx.x >> 5);
    const float4* xp = (const float4*)(x + (size_t)tok * DM);
    float s = 0.0f;
#pragma unroll
    for (int j = 0; j < 8; ++j) {
        float4 v = xp[lane + 32 * j];
        s += v.x * v.x + v.y * v.y + v.z * v.z + v.w * v.w;
    }
#pragma unroll
    for (int m = 16; m >= 1; m >>= 1) s += __shfl_xor(s, m, 32);
    // sqrt(c)=1: norm = min(||x||, 1/sqrt(c)-eps), arg = min(norm, 1-eps)
    float nrm = fminf(sqrtf(s), 1.0f - EPSV);
    float arg = fminf(nrm, 1.0f - EPSV);
    float dn  = (2.0f * atanhf(arg)) / (1.0f + EPSV);
    float hid = fmaxf(dn * w1[lane] + b1[lane], 0.0f);
    float c   = hid * w2[lane];
#pragma unroll
    for (int m = 16; m >= 1; m >>= 1) c += __shfl_xor(c, m, 32);
    if (lane == 0) {
        float r = sigmoidf_(c + b2[0]);
        float g = sigmoidf_(gr[tok] * pw[0] + gi[tok] * pw[1] + pb[0]);
        rw[tok] = r * g;
    }
}

// ---------------------------------------------------------------------------
// K2: A = A_low(64x32) @ A_high(32x64) and powers A^1..A^16 (for chunked scan).
__global__ void k_Apow(const float* __restrict__ Alow, const float* __restrict__ Ahigh,
                       float* __restrict__ A, float* __restrict__ Pw) {
    __shared__ float sA[DS * DS];
    __shared__ float sP[DS * DS];
    const int tid = threadIdx.x;  // 256 threads
#pragma unroll
    for (int i = 0; i < 16; ++i) {
        int e = tid + i * 256;
        int n = e >> 6, m = e & 63;
        float acc = 0.0f;
#pragma unroll
        for (int k = 0; k < 32; ++k) acc += Alow[n * 32 + k] * Ahigh[k * 64 + m];
        sA[e] = acc; sP[e] = acc;
        A[e] = acc; Pw[e] = acc;  // Pw[0] = A^1
    }
    __syncthreads();
    for (int j = 1; j < TCH; ++j) {
        float tmp[16];
#pragma unroll
        for (int i = 0; i < 16; ++i) {
            int e = tid + i * 256;
            int n = e >> 6, m = e & 63;
            float acc = 0.0f;
#pragma unroll
            for (int k = 0; k < 64; ++k) acc += sA[n * 64 + k] * sP[k * 64 + m];
            tmp[i] = acc;
        }
        __syncthreads();
#pragma unroll
        for (int i = 0; i < 16; ++i) {
            int e = tid + i * 256;
            sP[e] = tmp[i];
            Pw[(size_t)j * DS * DS + e] = tmp[i];  // Pw[j] = A^(j+1)
        }
        __syncthreads();
    }
}

// ---------------------------------------------------------------------------
// K3: u_scaled = (x @ B_w^T + B_b) * rank_weights, via v_wmma_f32_16x16x32_f16.
// All 4 waves of a block share one 16x1024 x-tile: TDM-stage it into LDS once,
// then feed WMMA A-fragments from LDS. B_w tiles stream from global (L2-hot).
__global__ void k_u_gemm(const float* __restrict__ x, const float* __restrict__ Bw,
                         const float* __restrict__ Bb, const float* __restrict__ rw,
                         float* __restrict__ u) {
    __shared__ __align__(16) float sx[16 * DM];   // 64 KB staged x tile
    const int wib  = threadIdx.x >> 5;
    const int wave = blockIdx.x * 4 + wib;
    const int mt = wave >> 2, nt = wave & 3;
    const int t0 = mt * 16, n0 = nt * 16;         // t0 identical for all 4 waves
    if (wib == 0) {
        tdm_load_1d(x + (size_t)t0 * DM, 0u, 16 * DM);
        __builtin_amdgcn_s_wait_tensorcnt(0);
    }
    __syncthreads();
    lds_acquire(sx);   // make the TDM write visible to the compiler
    v8f acc = {};
    for (int k0 = 0; k0 < DM; k0 += 32) {
        v16h a = load_a_frag_lds(sx, DM, k0);
        v16h b = load_b_frag(Bw + (size_t)n0 * DM, DM, k0);
        acc = __builtin_amdgcn_wmma_f32_16x16x32_f16(false, a, false, b,
                                                     (short)0, acc, false, false);
    }
    const int lane = threadIdx.x & 31;
    const int n = lane & 15, mb = (lane >> 4) * 8;
    const float bb = Bb[n0 + n];
#pragma unroll
    for (int v = 0; v < 8; ++v) {
        int t = t0 + mb + v;
        u[(size_t)t * DS + n0 + n] = (acc[v] + bb) * rw[t];
    }
}

// ---------------------------------------------------------------------------
// K4: parallel local scans: local_j = A * local_{j-1} + u_j (local_{-1}=0),
// chunk length 16. 4 chunks per block (64 threads each), 512 chunks total.
__global__ void k_localscan(const float* __restrict__ A, const float* __restrict__ u,
                            float* __restrict__ loc) {
    __shared__ float sh[4][DS];
    const int cl = threadIdx.x >> 6;
    const int n  = threadIdx.x & 63;
    const int cg = blockIdx.x * 4 + cl;      // flat chunk id (b*128 + c)
    const int t0 = cg * TCH;                 // flat token id of chunk start
    float Ar[DS];
    const float4* A4 = (const float4*)(A + (size_t)n * DS);
#pragma unroll
    for (int q = 0; q < 16; ++q) {
        float4 v = A4[q];
        Ar[4*q] = v.x; Ar[4*q+1] = v.y; Ar[4*q+2] = v.z; Ar[4*q+3] = v.w;
    }
    float hv = u[(size_t)t0 * DS + n];
    loc[(size_t)t0 * DS + n] = hv;
    sh[cl][n] = hv;
    __syncthreads();
    for (int j = 1; j < TCH; ++j) {
        float acc = u[(size_t)(t0 + j) * DS + n];
#pragma unroll
        for (int m = 0; m < DS; ++m) acc += Ar[m] * sh[cl][m];
        __syncthreads();
        sh[cl][n] = acc;
        loc[(size_t)(t0 + j) * DS + n] = acc;
        __syncthreads();
    }
}

// ---------------------------------------------------------------------------
// K5: sequential carry over chunks: hin(c+1) = A^16 * hin(c) + local_end(c).
// 4 batches in one block (64 threads each); only 128-deep dependency chain.
__global__ void k_carry(const float* __restrict__ Pw, const float* __restrict__ loc,
                        float* __restrict__ hin) {
    __shared__ float sh[RB][DS];
    const int b = threadIdx.x >> 6;
    const int n = threadIdx.x & 63;
    float PT[DS];
    const float4* P4 = (const float4*)(Pw + (size_t)(TCH - 1) * DS * DS + (size_t)n * DS);
#pragma unroll
    for (int q = 0; q < 16; ++q) {
        float4 v = P4[q];
        PT[4*q] = v.x; PT[4*q+1] = v.y; PT[4*q+2] = v.z; PT[4*q+3] = v.w;
    }
    sh[b][n] = 0.0f;
    __syncthreads();
    for (int c = 0; c < NCH; ++c) {
        hin[((size_t)(b * NCH + c)) * DS + n] = sh[b][n];
        int tl = b * SS + c * TCH + (TCH - 1);
        float acc = loc[(size_t)tl * DS + n];
#pragma unroll
        for (int m = 0; m < DS; ++m) acc += PT[m] * sh[b][m];
        __syncthreads();
        sh[b][n] = acc;
        __syncthreads();
    }
}

// ---------------------------------------------------------------------------
// K6: fix-up: hs[c*16+j] = local_j + A^(j+1) * hin(c). 4 tokens per block.
__global__ void k_fixup(const float* __restrict__ Pw, const float* __restrict__ hin,
                        float* __restrict__ loc) {
    __shared__ float sh[4][DS];
    const int tl = threadIdx.x >> 6;
    const int n  = threadIdx.x & 63;
    const int t  = blockIdx.x * 4 + tl;      // flat token
    const int cg = t >> 4;                   // flat chunk
    const int j  = t & 15;
    sh[tl][n] = hin[(size_t)cg * DS + n];
    __syncthreads();
    float Pr[DS];
    const float4* P4 = (const float4*)(Pw + (size_t)j * DS * DS + (size_t)n * DS);
#pragma unroll
    for (int q = 0; q < 16; ++q) {
        float4 v = P4[q];
        Pr[4*q] = v.x; Pr[4*q+1] = v.y; Pr[4*q+2] = v.z; Pr[4*q+3] = v.w;
    }
    float acc = loc[(size_t)t * DS + n];
#pragma unroll
    for (int m = 0; m < DS; ++m) acc += Pr[m] * sh[tl][m];
    loc[(size_t)t * DS + n] = acc;
}

// ---------------------------------------------------------------------------
// K7: y = hs @ C_w^T + C_b + D * x, via v_wmma_f32_16x16x32_f16.
// Block covers 1 M-tile x 4 N-tiles: TDM-stage the shared 16x64 hs tile and the
// block's 64 contiguous C_w rows (16 KB) into LDS; WMMA reads both from LDS.
__global__ void k_y_gemm(const float* __restrict__ hs, const float* __restrict__ Cw,
                         const float* __restrict__ Cb, const float* __restrict__ Dd,
                         const float* __restrict__ x, float* __restrict__ y) {
    __shared__ __align__(16) float sW[64 * DS + 16 * DS];  // C_w rows | hs tile
    const int wib  = threadIdx.x >> 5;
    const int wave = blockIdx.x * 4 + wib;
    const int mt = wave >> 6, dt = wave & 63;
    const int t0 = mt * 16, d0 = dt * 16;
    const int dtb = (blockIdx.x * 4) & 63;   // block's base N-tile
    if (wib == 0) {
        tdm_load_1d(Cw + (size_t)dtb * 16 * DS, 0u, 64 * DS);           // 16 KB
        tdm_load_1d(hs + (size_t)t0 * DS, (unsigned)(64 * DS * 4), 16 * DS); // 4 KB
        __builtin_amdgcn_s_wait_tensorcnt(0);
    }
    __syncthreads();
    lds_acquire(sW);   // make the TDM writes visible to the compiler
    v8f acc = {};
#pragma unroll
    for (int k0 = 0; k0 < DS; k0 += 32) {
        v16h a = load_a_frag_lds(sW + 64 * DS, DS, k0);
        v16h b = load_b_frag_lds(sW + wib * 16 * DS, DS, k0);
        acc = __builtin_amdgcn_wmma_f32_16x16x32_f16(false, a, false, b,
                                                     (short)0, acc, false, false);
    }
    const int lane = threadIdx.x & 31;
    const int n = lane & 15, mb = (lane >> 4) * 8;
    const int d = d0 + n;
    const float cb = Cb[d], dv = Dd[d];
#pragma unroll
    for (int v = 0; v < 8; ++v) {
        int t = t0 + mb + v;
        y[(size_t)t * DM + d] = acc[v] + cb + dv * x[(size_t)t * DM + d];
    }
}

// ---------------------------------------------------------------------------
extern "C" void kernel_launch(void* const* d_in, const int* in_sizes, int n_in,
                              void* d_out, int out_size, void* d_ws, size_t ws_size,
                              hipStream_t stream) {
    const float* x     = (const float*)d_in[0];
    const float* gr    = (const float*)d_in[1];
    const float* gi    = (const float*)d_in[2];
    const float* Alow  = (const float*)d_in[3];
    const float* Ahigh = (const float*)d_in[4];
    const float* Bw    = (const float*)d_in[5];
    const float* Bb    = (const float*)d_in[6];
    const float* Cw    = (const float*)d_in[7];
    const float* Cb    = (const float*)d_in[8];
    const float* Dd    = (const float*)d_in[9];
    const float* w1    = (const float*)d_in[10];
    const float* b1    = (const float*)d_in[11];
    const float* w2    = (const float*)d_in[12];
    const float* b2    = (const float*)d_in[13];
    const float* pw    = (const float*)d_in[14];
    const float* pb    = (const float*)d_in[15];
    float* y  = (float*)d_out;
    float* ws = (float*)d_ws;

    // workspace layout (floats), all 64B-aligned offsets
    float* rw  = ws;               // 8192     rank weights
    float* A   = ws + 8192;        // 4096     A (64x64)
    float* Pw  = ws + 12288;       // 65536    A^1..A^16
    float* u   = ws + 77824;       // 524288   u_scaled [8192][64]
    float* loc = ws + 602112;      // 524288   local scans -> hs (in place)
    float* hin = ws + 1126400;     // 32768    chunk carry-in states

    k_rankw    <<<TOK / 8, 256, 0, stream>>>(x, gr, gi, w1, b1, w2, b2, pw, pb, rw);
    k_Apow     <<<1, 256, 0, stream>>>(Alow, Ahigh, A, Pw);
    k_u_gemm   <<<(TOK / 16) * (DS / 16) / 4, 128, 0, stream>>>(x, Bw, Bb, rw, u);
    k_localscan<<<(RB * NCH) / 4, 256, 0, stream>>>(A, u, loc);
    k_carry    <<<1, 256, 0, stream>>>(Pw, loc, hin);
    k_fixup    <<<TOK / 4, 256, 0, stream>>>(Pw, hin, loc);
    k_y_gemm   <<<(TOK / 16) * (DM / 16) / 4, 128, 0, stream>>>(loc, Cw, Cb, Dd, x, y);
}
